// VariableSelectionNetwork_13735305412895
// MI455X (gfx1250) — matrix operand order repaired and checked by reading, not compiled, not run on hardware
//
#include <hip/hip_runtime.h>
#include <hip/hip_bf16.h>

#define B_  16
#define T_  256
#define V_  64
#define H_  256
#define BT_ (B_ * T_)          // 4096 rows
#define EPSV 1e-5f

typedef _Float16 v16h __attribute__((ext_vector_type(16)));
typedef _Float16 v8h  __attribute__((ext_vector_type(8)));
typedef float    v8f  __attribute__((ext_vector_type(8)));

__device__ __forceinline__ float eluf(float x)  { return x > 0.f ? x : (__expf(x) - 1.f); }
__device__ __forceinline__ float sigmf(float x) { return 1.f / (1.f + __expf(-x)); }

__device__ __forceinline__ v8f wmma_f16(v16h a, v16h b, v8f c) {
    // D = A(16x32 f16) * B(32x16 f16) + C(16x16 f32)
    return __builtin_amdgcn_wmma_f32_16x16x32_f16(false, a, false, b, (short)0, c, false, false);
}

// A fragment: lane<16 -> M=lane, K = kb..kb+7 and kb+16..kb+23 (kb = kk*32)
//             lane>=16 -> M=lane-16, kb offset by +8.  p points at A_lds[m*H + kbase].
__device__ __forceinline__ v16h load_a_frag(const _Float16* p) {
    v8h a0 = *(const v8h*)p;
    v8h a1 = *(const v8h*)(p + 16);
    v16h r;
#pragma unroll
    for (int i = 0; i < 8; ++i) { r[i] = a0[i]; r[i + 8] = a1[i]; }
    return r;
}

// B fragment: packed layout gives 16 contiguous halves per lane.
__device__ __forceinline__ v16h load_b_frag(const _Float16* p) {
    v8h a0 = *(const v8h*)p;
    v8h a1 = *(const v8h*)(p + 8);
    v16h r;
#pragma unroll
    for (int i = 0; i < 8; ++i) { r[i] = a0[i]; r[i + 8] = a1[i]; }
    return r;
}

// ---------------------------------------------------------------------------
// Kernel 0: repack W2 [V][H][H] f32 -> f16 fragment layout
// W2p[v][ntg(16)][kk(8)][lane(32)][j(16)]  where
//   n = ntg*16 + (lane&15),  k = kk*32 + (lane>>4)*8 + (j>>3)*16 + (j&7)
// ---------------------------------------------------------------------------
__global__ void pack_w2(const float* __restrict__ W2, _Float16* __restrict__ W2p) {
    int idx = (blockIdx.x * blockDim.x + threadIdx.x) * 8;   // 8 halves / thread
    int j0   =  idx        & 15;
    int lane = (idx >> 4)  & 31;
    int kk   = (idx >> 9)  & 7;
    int ntg  = (idx >> 12) & 15;
    int v    =  idx >> 16;
    int n    = ntg * 16 + (lane & 15);
    int kb   = kk * 32 + (lane >> 4) * 8 + (j0 >> 3) * 16;
    _Float16 tmp[8];
#pragma unroll
    for (int j = 0; j < 8; ++j)
        tmp[j] = (_Float16)W2[((size_t)(v * H_ + (kb + j))) * H_ + n];
    *(v8h*)(W2p + idx) = *(const v8h*)tmp;
}

// ---------------------------------------------------------------------------
// Kernel 1: weight network GRN (V->V) + layernorm + softmax -> Wts[BT][V]
// 64 threads per row, 4 rows per block.
// ---------------------------------------------------------------------------
__global__ void vsn_weights(const float* __restrict__ x,
                            const float* __restrict__ nW1, const float* __restrict__ nb1,
                            const float* __restrict__ nW2, const float* __restrict__ nb2,
                            const float* __restrict__ nWg, const float* __restrict__ nbg,
                            const float* __restrict__ ng,  const float* __restrict__ nbe,
                            float* __restrict__ Wts) {
    __shared__ float sx[4][V_];
    __shared__ float se[4][V_];
    __shared__ float sp[4][V_];
    const int t = threadIdx.x;
    const int r = t >> 6;
    const int k = t & 63;
    const int row = blockIdx.x * 4 + r;

    sx[r][k] = x[row * V_ + k];
    __syncthreads();

    float h1 = nb1[k], hg = nbg[k];
#pragma unroll 8
    for (int v = 0; v < V_; ++v) {
        float xv = sx[r][v];
        h1 += xv * nW1[v * V_ + k];
        hg += xv * nWg[v * V_ + k];
    }
    se[r][k] = eluf(h1);
    float wg = sigmf(hg);
    __syncthreads();

    float wres = nb2[k];
#pragma unroll 8
    for (int v = 0; v < V_; ++v) wres += se[r][v] * nW2[v * V_ + k];
    float pre = sx[r][k] + wg * wres;
    sp[r][k] = pre;
    __syncthreads();

    float mu = 0.f;
    for (int i = 0; i < V_; ++i) mu += sp[r][i];
    mu *= (1.f / V_);
    float var = 0.f;
    for (int i = 0; i < V_; ++i) { float d = sp[r][i] - mu; var += d * d; }
    var *= (1.f / V_);
    float wout = (pre - mu) * rsqrtf(var + EPSV) * ng[k] + nbe[k];
    se[r][k] = wout;
    __syncthreads();

    float mx = -3.0e38f;
    for (int i = 0; i < V_; ++i) mx = fmaxf(mx, se[r][i]);
    float e = __expf(wout - mx);
    sp[r][k] = e;
    __syncthreads();
    float s = 0.f;
    for (int i = 0; i < V_; ++i) s += sp[r][i];
    Wts[row * V_ + k] = e / s;
}

// ---------------------------------------------------------------------------
// Kernel 2: per-variable GRN GEMM (WMMA) + fused LN + weighted combine.
// Block = 32 rows, 256 threads = 8 waves; wave w owns columns [w*32, w*32+32).
// ---------------------------------------------------------------------------
__launch_bounds__(256)
__global__ void vsn_main(const float* __restrict__ x,
                         const float* __restrict__ W1, const float* __restrict__ b1,
                         const float* __restrict__ b2,
                         const float* __restrict__ Wg, const float* __restrict__ bg,
                         const float* __restrict__ Wsk, const float* __restrict__ bs,
                         const float* __restrict__ g1, const float* __restrict__ be1,
                         const _Float16* __restrict__ W2p,
                         const float* __restrict__ Wts,
                         float* __restrict__ out) {
    __shared__ _Float16 A_lds[32 * H_];     // 16 KB f16 A tile
    __shared__ float xv[32], wrow[32], s_sum[32], s_ssq[32];

    const int t    = threadIdx.x;
    const int lane = t & 31;
    const int w    = t >> 5;        // wave 0..7
    const int hl   = lane >> 4;     // half-lane group
    const int lm   = lane & 15;
    const int row0 = blockIdx.x * 32;

    const int n0  = w * 32 + lm;    // column of tile nt=0
    const int n1  = n0 + 16;        // column of tile nt=1

    v8f comb[2][2];
#pragma unroll
    for (int mt = 0; mt < 2; ++mt)
#pragma unroll
        for (int nt = 0; nt < 2; ++nt)
#pragma unroll
            for (int i = 0; i < 8; ++i) comb[mt][nt][i] = 0.f;

    for (int v = 0; v < V_; ++v) {
        __syncthreads();
        if (t < 32) {
            xv[t]    = x[(row0 + t) * V_ + v];
            wrow[t]  = Wts[(row0 + t) * V_ + v];
            s_sum[t] = 0.f;
            s_ssq[t] = 0.f;
        }
        {   // build f16 A tile: a[m][h] = elu(x[m,v]*W1[v,h] + b1[v,h])
            int m  = t >> 3;
            int h0 = (t & 7) * 32;
            float xm = x[(row0 + m) * V_ + v];
            const float* w1p = W1 + v * H_ + h0;
            const float* b1p = b1 + v * H_ + h0;
#pragma unroll 8
            for (int i = 0; i < 32; ++i)
                A_lds[m * H_ + h0 + i] = (_Float16)eluf(xm * w1p[i] + b1p[i]);
        }
        __syncthreads();

        v8f acc[2][2];
#pragma unroll
        for (int mt = 0; mt < 2; ++mt)
#pragma unroll
            for (int nt = 0; nt < 2; ++nt)
#pragma unroll
                for (int i = 0; i < 8; ++i) acc[mt][nt][i] = 0.f;

        const _Float16* bbase = W2p + (size_t)v * (16 * 8 * 32 * 16);
#pragma unroll
        for (int kk = 0; kk < 8; ++kk) {
            int kb = kk * 32 + hl * 8;
            v16h a0 = load_a_frag(A_lds + (lm      ) * H_ + kb);
            v16h a1 = load_a_frag(A_lds + (lm + 16 ) * H_ + kb);
            v16h bf0 = load_b_frag(bbase + (size_t)(((w * 2 + 0) * 8 + kk) * 32 + lane) * 16);
            v16h bf1 = load_b_frag(bbase + (size_t)(((w * 2 + 1) * 8 + kk) * 32 + lane) * 16);
            acc[0][0] = wmma_f16(a0, bf0, acc[0][0]);
            acc[0][1] = wmma_f16(a0, bf1, acc[0][1]);
            acc[1][0] = wmma_f16(a1, bf0, acc[1][0]);
            acc[1][1] = wmma_f16(a1, bf1, acc[1][1]);
        }

        // ---- epilogue: pre = skip + gate*(res + b2), LN stats over H ----
        const int pi0 = v * H_ + n0, pi1 = v * H_ + n1;
        const float wg0 = Wg[pi0], bg0 = bg[pi0], ws0 = Wsk[pi0], bs0 = bs[pi0], b20 = b2[pi0];
        const float wg1 = Wg[pi1], bg1 = bg[pi1], ws1 = Wsk[pi1], bs1 = bs[pi1], b21 = b2[pi1];

        float s0[2][8], s1[2][8];
#pragma unroll
        for (int mt = 0; mt < 2; ++mt)
#pragma unroll
            for (int rr = 0; rr < 8; ++rr) {
                int m = mt * 16 + (hl ? 8 + rr : rr);
                float xm = xv[m];
                float p0 = (xm * ws0 + bs0) + sigmf(xm * wg0 + bg0) * (acc[mt][0][rr] + b20);
                float p1 = (xm * ws1 + bs1) + sigmf(xm * wg1 + bg1) * (acc[mt][1][rr] + b21);
                acc[mt][0][rr] = p0;
                acc[mt][1][rr] = p1;
                s0[mt][rr] = p0 + p1;
                s1[mt][rr] = p0 * p0 + p1 * p1;
            }
        // reduce over the 16 lanes sharing a row
#pragma unroll
        for (int mk = 1; mk < 16; mk <<= 1) {
#pragma unroll
            for (int mt = 0; mt < 2; ++mt)
#pragma unroll
                for (int rr = 0; rr < 8; ++rr) {
                    s0[mt][rr] += __shfl_xor(s0[mt][rr], mk, 32);
                    s1[mt][rr] += __shfl_xor(s1[mt][rr], mk, 32);
                }
        }
        if (lm == 0) {
#pragma unroll
            for (int mt = 0; mt < 2; ++mt)
#pragma unroll
                for (int rr = 0; rr < 8; ++rr) {
                    int m = mt * 16 + (hl ? 8 + rr : rr);
                    atomicAdd(&s_sum[m], s0[mt][rr]);
                    atomicAdd(&s_ssq[m], s1[mt][rr]);
                }
        }
        __syncthreads();

        const float g1c0 = g1[pi0], bec0 = be1[pi0];
        const float g1c1 = g1[pi1], bec1 = be1[pi1];
#pragma unroll
        for (int mt = 0; mt < 2; ++mt)
#pragma unroll
            for (int rr = 0; rr < 8; ++rr) {
                int m = mt * 16 + (hl ? 8 + rr : rr);
                float mu  = s_sum[m] * (1.f / H_);
                float var = s_ssq[m] * (1.f / H_) - mu * mu;
                float rs  = rsqrtf(var + EPSV);
                float wv  = wrow[m];
                comb[mt][0][rr] += wv * ((acc[mt][0][rr] - mu) * rs * g1c0 + bec0);
                comb[mt][1][rr] += wv * ((acc[mt][1][rr] - mu) * rs * g1c1 + bec1);
            }
    }

#pragma unroll
    for (int mt = 0; mt < 2; ++mt)
#pragma unroll
        for (int rr = 0; rr < 8; ++rr) {
            int m = mt * 16 + (hl ? 8 + rr : rr);
            out[(row0 + m) * H_ + n0] = comb[mt][0][rr];
            out[(row0 + m) * H_ + n1] = comb[mt][1][rr];
        }
}

extern "C" void kernel_launch(void* const* d_in, const int* in_sizes, int n_in,
                              void* d_out, int out_size, void* d_ws, size_t ws_size,
                              hipStream_t stream) {
    const float* x   = (const float*)d_in[0];
    const float* W1  = (const float*)d_in[1];
    const float* b1  = (const float*)d_in[2];
    const float* W2  = (const float*)d_in[3];
    const float* b2  = (const float*)d_in[4];
    const float* Wg  = (const float*)d_in[5];
    const float* bg  = (const float*)d_in[6];
    const float* Wsk = (const float*)d_in[7];
    const float* bs  = (const float*)d_in[8];
    const float* g1  = (const float*)d_in[9];
    const float* be1 = (const float*)d_in[10];
    const float* nW1 = (const float*)d_in[11];
    const float* nb1 = (const float*)d_in[12];
    const float* nW2 = (const float*)d_in[13];
    const float* nb2 = (const float*)d_in[14];
    const float* nWg = (const float*)d_in[15];
    const float* nbg = (const float*)d_in[16];
    const float* ng  = (const float*)d_in[17];
    const float* nbe = (const float*)d_in[18];

    _Float16* W2p = (_Float16*)d_ws;                                  // 8.39 MB
    float* Wts = (float*)((char*)d_ws +
                          (size_t)V_ * 16 * 8 * 32 * 16 * sizeof(_Float16)); // + 1 MB

    pack_w2    <<<2048,      256, 0, stream>>>(W2, W2p);
    vsn_weights<<<BT_ / 4,   256, 0, stream>>>(x, nW1, nb1, nW2, nb2, nWg, nbg, ng, nbe, Wts);
    vsn_main   <<<BT_ / 32,  256, 0, stream>>>(x, W1, b1, b2, Wg, bg, Wsk, bs, g1, be1,
                                               W2p, Wts, (float*)d_out);
}
